// TTGPT_82136954568965
// MI455X (gfx1250) — matrix operand order repaired and checked by reading, not compiled, not run on hardware
//
#include <hip/hip_runtime.h>
#include <hip/hip_bf16.h>
#include <math.h>

// ---------------- types ----------------
typedef __attribute__((ext_vector_type(16))) __bf16        v16bf;
typedef __attribute__((ext_vector_type(8)))  float         v8f;
typedef __attribute__((ext_vector_type(8)))  unsigned int  v8u;
typedef __attribute__((ext_vector_type(4)))  unsigned int  u32x4;
typedef __attribute__((ext_vector_type(4)))  float         f32x4;
typedef __attribute__((ext_vector_type(2)))  float         f32x2;

#define NL   12
#define BSZ  4
#define TT   1024
#define DM   1024
#define NH   16
#define HDM  64
#define NV   50257
#define FF   4096

// ---------------- helpers ----------------
__device__ __forceinline__ unsigned short f2bf(float f) {
  unsigned int u = __builtin_bit_cast(unsigned int, f);
  u += 0x7FFFu + ((u >> 16) & 1u);   // round-to-nearest-even
  return (unsigned short)(u >> 16);
}
__device__ __forceinline__ unsigned int pack2bf(float lo, float hi) {
  return (unsigned int)f2bf(lo) | ((unsigned int)f2bf(hi) << 16);
}

// A-fragment (16x32 bf16): per-lane dword offsets = v + (laneHi?8:0)
// -> two contiguous b128 loads per lane.
__device__ __forceinline__ v8u load_a_frag(const unsigned int* base_dw, int lane) {
  const int hi4 = (lane & 16) ? 8 : 0;
  u32x4 a0 = *(const u32x4*)(base_dw + hi4);
  u32x4 a1 = *(const u32x4*)(base_dw + hi4 + 4);
  v8u au;
  #pragma unroll
  for (int i = 0; i < 4; ++i) { au[i] = a0[i]; au[i + 4] = a1[i]; }
  return au;
}
// B-fragment (32x16 bf16) from raw row-major source (row = N index, cols = K):
// per-lane dword offsets = v + (laneHi?8:0) -> two contiguous b128 loads.
__device__ __forceinline__ v8u load_b_frag_raw(const unsigned int* base_dw, int lane) {
  const int hi8 = (lane & 16) ? 8 : 0;
  u32x4 b0 = *(const u32x4*)(base_dw + hi8);
  u32x4 b1 = *(const u32x4*)(base_dw + hi8 + 4);
  v8u bu;
  #pragma unroll
  for (int i = 0; i < 4; ++i) { bu[i] = b0[i]; bu[i + 4] = b1[i]; }
  return bu;
}
// B-fragment from pre-packed weights: 8 contiguous dwords per lane.
__device__ __forceinline__ v8u load_b_frag_packed(const unsigned int* bp) {
  u32x4 b0 = *(const u32x4*)bp;
  u32x4 b1 = *(const u32x4*)(bp + 4);
  v8u bu;
  #pragma unroll
  for (int i = 0; i < 4; ++i) { bu[i] = b0[i]; bu[i + 4] = b1[i]; }
  return bu;
}
// pack target layout: lane (l&15)=N col, k = 2*v + (laneHi?16:0)
__device__ __forceinline__ int b_koff(int v, int lane) {
  return 2 * v + ((lane & 16) ? 16 : 0);
}

__device__ __forceinline__ v8f do_wmma(const v8u& au, const v8u& bu, v8f c) {
  v16bf a = __builtin_bit_cast(v16bf, au);
  v16bf b = __builtin_bit_cast(v16bf, bu);
  return __builtin_amdgcn_wmma_f32_16x16x32_bf16(false, a, false, b, (short)0, c,
                                                 false, false);
}

// ---------------- embedding (float4) ----------------
__global__ void embed_kernel(const int* __restrict__ idx,
                             const float* __restrict__ wte,
                             const float* __restrict__ wpe,
                             float* __restrict__ x) {
  int i = blockIdx.x * blockDim.x + threadIdx.x;          // over (B*T*D)/4
  if (i >= BSZ * TT * DM / 4) return;
  int dq = i % (DM / 4);
  int bt = i / (DM / 4);
  int t  = bt % TT;
  f32x4 a = *(const f32x4*)(wte + (size_t)idx[bt] * DM + dq * 4);
  f32x4 p = *(const f32x4*)(wpe + (size_t)t * DM + dq * 4);
  *(f32x4*)(x + (size_t)bt * DM + dq * 4) = a + p;
}

// ---------------- gamma-only layernorm: f32 in -> bf16 out ----------------
__global__ void ln_kernel(const float* __restrict__ x,
                          const float* __restrict__ g,
                          unsigned short* __restrict__ out, int D) {
  __shared__ float red[8];
  int row = blockIdx.x;
  const float* xr = x + (size_t)row * D;
  int lane = threadIdx.x & 31, w = threadIdx.x >> 5, nw = blockDim.x >> 5;

  float s = 0.f;
  for (int d = threadIdx.x * 2; d < D; d += blockDim.x * 2) {
    f32x2 v = *(const f32x2*)(xr + d);
    s += v[0] + v[1];
  }
  #pragma unroll
  for (int m = 16; m; m >>= 1) s += __shfl_xor(s, m, 32);
  if (lane == 0) red[w] = s;
  __syncthreads();
  float mean = 0.f;
  for (int i = 0; i < nw; ++i) mean += red[i];
  mean /= (float)D;
  __syncthreads();

  float vs = 0.f;
  for (int d = threadIdx.x * 2; d < D; d += blockDim.x * 2) {
    f32x2 v = *(const f32x2*)(xr + d);
    float d0 = v[0] - mean, d1 = v[1] - mean;
    vs += d0 * d0 + d1 * d1;
  }
  #pragma unroll
  for (int m = 16; m; m >>= 1) vs += __shfl_xor(vs, m, 32);
  if (lane == 0) red[w] = vs;
  __syncthreads();
  float var = 0.f;
  for (int i = 0; i < nw; ++i) var += red[i];
  float rs = rsqrtf(var / (float)D + 1e-5f);

  unsigned int* od = (unsigned int*)(out + (size_t)row * D);
  for (int d = threadIdx.x * 2; d < D; d += blockDim.x * 2) {
    f32x2 v = *(const f32x2*)(xr + d);
    od[d >> 1] = pack2bf((v[0] - mean) * rs * g[d], (v[1] - mean) * rs * g[d + 1]);
  }
}

// ---------------- weight pack: f32 W[K,N] -> per-lane WMMA B fragments ------
__global__ void pack_kernel(const float* __restrict__ W,
                            unsigned int* __restrict__ P, int K, int N) {
  int g    = (blockIdx.x * blockDim.x + threadIdx.x) >> 5;
  int lane = threadIdx.x & 31;
  int KT = K >> 5, NT = N >> 4;
  if (g >= KT * NT) return;
  int nt = g / KT, kt = g % KT;
  int n  = nt * 16 + (lane & 15);
  u32x4 d0, d1;
  #pragma unroll
  for (int v = 0; v < 8; ++v) {
    int k = kt * 32 + b_koff(v, lane);
    unsigned int pk = pack2bf(W[(size_t)k * N + n], W[(size_t)(k + 1) * N + n]);
    if (v < 4) d0[v] = pk; else d1[v - 4] = pk;
  }
  unsigned int* dst = P + (size_t)g * 256 + lane * 8;
  *(u32x4*)dst       = d0;
  *(u32x4*)(dst + 4) = d1;
}

// ---------------- register-blocked WMMA GEMM: 32x64 output per wave --------
// C[32,64] block = A(bf16,[M,K]) @ Wpacked ; 2 M-frags x 4 N-frags, 8 WMMA/k-step
// EPI 0: f32 out = resid + C ;  EPI 1: bf16 out = C ;  EPI 2: bf16 out = gelu(C)
template <int EPI>
__global__ void gemm_wmma(const unsigned short* __restrict__ A,
                          const unsigned int* __restrict__ P,
                          void* __restrict__ out,
                          const float* __restrict__ resid,
                          int M, int N, int K) {
  const int lane = threadIdx.x & 31;
  const int wave = blockIdx.x * (blockDim.x >> 5) + (threadIdx.x >> 5);
  const int NB = N >> 6, MB = M >> 5, KT = K >> 5;
  if (wave >= MB * NB) return;                 // wave-uniform: EXEC stays all-1
  const int bm = wave / NB, bn = wave % NB;

  const unsigned int* arow[2];
  #pragma unroll
  for (int mi = 0; mi < 2; ++mi) {
    int am = (bm * 2 + mi) * 16 + (lane & 15);
    arow[mi] = (const unsigned int*)A + (((size_t)am * K) >> 1);
  }
  const unsigned int* bp[4];
  #pragma unroll
  for (int ni = 0; ni < 4; ++ni)
    bp[ni] = P + (size_t)(bn * 4 + ni) * KT * 256 + lane * 8;

  v8f c[2][4];
  #pragma unroll
  for (int mi = 0; mi < 2; ++mi)
    #pragma unroll
    for (int ni = 0; ni < 4; ++ni) c[mi][ni] = (v8f){};

  for (int kt = 0; kt < KT; ++kt) {
    v8u au[2], bu[4];
    #pragma unroll
    for (int mi = 0; mi < 2; ++mi) au[mi] = load_a_frag(arow[mi] + kt * 16, lane);
    #pragma unroll
    for (int ni = 0; ni < 4; ++ni) {
      bu[ni] = load_b_frag_packed(bp[ni]);
      bp[ni] += 256;
      __builtin_prefetch(bp[ni], 0, 1);        // global_prefetch next B tile
    }
    #pragma unroll
    for (int mi = 0; mi < 2; ++mi)
      #pragma unroll
      for (int ni = 0; ni < 4; ++ni)
        c[mi][ni] = do_wmma(au[mi], bu[ni], c[mi][ni]);
  }

  const int rbase = (lane & 16) ? 8 : 0;
  #pragma unroll
  for (int mi = 0; mi < 2; ++mi) {
    const int mbase = (bm * 2 + mi) * 16 + rbase;
    #pragma unroll
    for (int ni = 0; ni < 4; ++ni) {
      const int cn = (bn * 4 + ni) * 16 + (lane & 15);
      #pragma unroll
      for (int v = 0; v < 8; ++v) {
        size_t o   = (size_t)(mbase + v) * N + cn;
        float  val = c[mi][ni][v];
        if constexpr (EPI == 0) {
          ((float*)out)[o] = resid[o] + val;
        } else if constexpr (EPI == 1) {
          ((unsigned short*)out)[o] = f2bf(val);
        } else {
          float ge = 0.5f * val * (1.0f + erff(val * 0.70710678118654752f));
          ((unsigned short*)out)[o] = f2bf(ge);
        }
      }
    }
  }
}

// ---------------- V transpose: v[B*T, D] -> vt[B,H,HD,T] (bf16, dword out) --
__global__ void vtrans_kernel(const unsigned short* __restrict__ v,
                              unsigned int* __restrict__ vt) {
  int j = blockIdx.x * blockDim.x + threadIdx.x;     // over (B*H*HD*T)/2
  if (j >= BSZ * NH * HDM * TT / 2) return;
  int t2  = j % (TT / 2);
  int rem = j / (TT / 2);
  int d   = rem % HDM;
  int bh  = rem / HDM;
  int b = bh / NH, h = bh % NH;
  size_t s0 = (size_t)(b * TT + t2 * 2) * DM + h * HDM + d;
  float lo = (float)__builtin_bit_cast(__hip_bfloat16, v[s0]);
  float hi = (float)__builtin_bit_cast(__hip_bfloat16, v[s0 + DM]);
  vt[j] = pack2bf(lo, hi);
}

// ---------------- fused attention per (b,h, 16-query-row block) -------------
__global__ void attn_kernel(const unsigned short* __restrict__ q,
                            const unsigned short* __restrict__ k,
                            const unsigned short* __restrict__ vt,
                            unsigned short* __restrict__ cx) {
  extern __shared__ char smraw[];
  float*        sm = (float*)smraw;                          // 16*T f32 = 64KB
  unsigned int* pm = (unsigned int*)(smraw + 16 * TT * 4);   // 16*T bf16 = 32KB

  const int lane = threadIdx.x & 31;
  const int QT = TT / 16;
  const int qt = blockIdx.x % QT;
  const int bh = blockIdx.x / QT;
  const int b = bh / NH, h = bh % NH;

  const unsigned int* qd = (const unsigned int*)q;
  const unsigned int* kd = (const unsigned int*)k;

  const int arow = b * TT + qt * 16 + (lane & 15);
  const unsigned int* qbase = qd + (((size_t)arow * DM + h * HDM) >> 1);

  // ---- scores + scale + causal mask ----
  for (int jt = 0; jt < QT; ++jt) {
    v8f s = {};
    const int krow = b * TT + jt * 16 + (lane & 15);
    const unsigned int* kbase = kd + (((size_t)krow * DM + h * HDM) >> 1);
    #pragma unroll
    for (int kb = 0; kb < HDM; kb += 32) {
      v8u au = load_a_frag(qbase + (kb >> 1), lane);
      v8u bu = load_b_frag_raw(kbase + (kb >> 1), lane);
      s = do_wmma(au, bu, s);
    }
    const int j     = jt * 16 + (lane & 15);
    const int rbase = (lane & 16) ? 8 : 0;
    #pragma unroll
    for (int v = 0; v < 8; ++v) {
      int i = qt * 16 + rbase + v;                 // global query index
      sm[(rbase + v) * TT + j] = (j <= i) ? s[v] * 0.125f : -1e9f;
    }
  }
  __syncthreads();

  // ---- softmax per row (wave32 cooperative, paired element accesses) ----
  for (int r = 0; r < 16; ++r) {
    float* smr = sm + r * TT;
    float mx = -1e30f;
    for (int j = lane * 2; j < TT; j += 64) {
      f32x2 v = *(const f32x2*)(smr + j);
      mx = fmaxf(mx, fmaxf(v[0], v[1]));
    }
    #pragma unroll
    for (int m = 16; m; m >>= 1) mx = fmaxf(mx, __shfl_xor(mx, m, 32));
    float sum = 0.f;
    for (int j = lane * 2; j < TT; j += 64) {
      f32x2 v = *(const f32x2*)(smr + j);
      f32x2 e = { expf(v[0] - mx), expf(v[1] - mx) };
      *(f32x2*)(smr + j) = e;
      sum += e[0] + e[1];
    }
    #pragma unroll
    for (int m = 16; m; m >>= 1) sum += __shfl_xor(sum, m, 32);
    float inv = 1.0f / sum;
    unsigned int* pmr = pm + ((r * TT) >> 1);
    for (int j = lane * 2; j < TT; j += 64) {
      f32x2 v = *(const f32x2*)(smr + j);
      pmr[j >> 1] = pack2bf(v[0] * inv, v[1] * inv);
    }
  }
  __syncthreads();

  // ---- ctx = P @ V ----
  const unsigned int* vtd = (const unsigned int*)vt;
  const unsigned int* pdr = pm + (((lane & 15) * TT) >> 1);
  for (int nt2 = 0; nt2 < HDM / 16; ++nt2) {
    v8f c = {};
    const int n = nt2 * 16 + (lane & 15);
    const unsigned int* vrow = vtd + (((size_t)(bh * HDM + n) * TT) >> 1);
    for (int kt = 0; kt < TT / 32; ++kt) {
      v8u au = load_a_frag(pdr + kt * 16, lane);     // ds_load_b128 from LDS
      v8u bu = load_b_frag_raw(vrow + kt * 16, lane);
      c = do_wmma(au, bu, c);
    }
    const int col   = h * HDM + nt2 * 16 + (lane & 15);
    const int rbase = (lane & 16) ? 8 : 0;
    #pragma unroll
    for (int v = 0; v < 8; ++v) {
      int row = b * TT + qt * 16 + rbase + v;
      cx[(size_t)row * DM + col] = f2bf(c[v]);
    }
  }
}

// ---------------- lm_head: out[B,V] = hfinal(last pos) @ wte^T --------------
__global__ void lmhead_kernel(const unsigned short* __restrict__ hf,
                              const float* __restrict__ wte,
                              float* __restrict__ out) {
  const int lane = threadIdx.x & 31;
  const int NT   = (NV + 15) / 16;
  const int wave = blockIdx.x * (blockDim.x >> 5) + (threadIdx.x >> 5);
  if (wave >= NT) return;
  const int nt = wave;
  const int m  = lane & 15;
  const int n  = nt * 16 + (lane & 15);
  const int nn = (n < NV) ? n : 0;                 // clamp; zero later
  const unsigned int* hd = (const unsigned int*)hf;
  const unsigned int* abase =
      hd + ((((size_t)m * TT + (TT - 1)) * DM) >> 1);   // last position row
  const float* wrow = wte + (size_t)nn * DM;
  const v8u zu = {};

  v8f c = {};
  for (int kt = 0; kt < DM / 32; ++kt) {
    v8u au = (m < BSZ) ? load_a_frag(abase + kt * 16, lane) : zu;
    // B-fragment: 16 contiguous f32 from this lane's wte row -> 4x float4
    const int kb16 = kt * 32 + ((lane & 16) ? 16 : 0);
    f32x4 w0 = *(const f32x4*)(wrow + kb16);
    f32x4 w1 = *(const f32x4*)(wrow + kb16 + 4);
    f32x4 w2 = *(const f32x4*)(wrow + kb16 + 8);
    f32x4 w3 = *(const f32x4*)(wrow + kb16 + 12);
    v8u bu;
    bu[0] = pack2bf(w0[0], w0[1]); bu[1] = pack2bf(w0[2], w0[3]);
    bu[2] = pack2bf(w1[0], w1[1]); bu[3] = pack2bf(w1[2], w1[3]);
    bu[4] = pack2bf(w2[0], w2[1]); bu[5] = pack2bf(w2[2], w2[3]);
    bu[6] = pack2bf(w3[0], w3[1]); bu[7] = pack2bf(w3[2], w3[3]);
    if (n >= NV) bu = zu;
    c = do_wmma(au, bu, c);
  }
  const int rb = (lane & 16) ? 8 : 0;
  #pragma unroll
  for (int v = 0; v < 8; ++v) {
    int r = rb + v;
    if (r < BSZ && n < NV) out[(size_t)r * NV + n] = c[v];
  }
}

// ---------------- host orchestration ----------------
extern "C" void kernel_launch(void* const* d_in, const int* in_sizes, int n_in,
                              void* d_out, int out_size, void* d_ws, size_t ws_size,
                              hipStream_t stream) {
  (void)in_sizes; (void)n_in; (void)out_size; (void)ws_size;
  const int*   idx  = (const int*)d_in[0];
  const float* wte  = (const float*)d_in[1];
  const float* wpe  = (const float*)d_in[2];
  const float* ln1g = (const float*)d_in[3];
  const float* ln2g = (const float*)d_in[4];
  const float* wq   = (const float*)d_in[5];
  const float* wk   = (const float*)d_in[6];
  const float* wv   = (const float*)d_in[7];
  const float* wo   = (const float*)d_in[8];
  const float* wfc  = (const float*)d_in[9];
  const float* wpr  = (const float*)d_in[10];
  const float* lnfg = (const float*)d_in[11];
  float* out = (float*)d_out;

  const int M = BSZ * TT;  // 4096 rows

  char* ws = (char*)d_ws;
  size_t off = 0;
  auto alloc = [&](size_t bytes) -> char* {
    char* p = ws + off;
    off = (off + bytes + 255) & ~(size_t)255;
    return p;
  };
  float*          xf  = (float*)alloc((size_t)M * DM * 4);
  unsigned short* hb  = (unsigned short*)alloc((size_t)M * DM * 2);
  unsigned short* qb  = (unsigned short*)alloc((size_t)M * DM * 2);
  unsigned short* kb  = (unsigned short*)alloc((size_t)M * DM * 2);
  unsigned short* vb  = (unsigned short*)alloc((size_t)M * DM * 2);
  unsigned int*   vt  = (unsigned int*)alloc((size_t)M * DM * 2);
  unsigned short* cxb = (unsigned short*)alloc((size_t)M * DM * 2);
  unsigned short* fcb = (unsigned short*)alloc((size_t)M * FF * 2);
  unsigned int*   pq  = (unsigned int*)alloc((size_t)DM * DM * 2);
  unsigned int*   pk  = (unsigned int*)alloc((size_t)DM * DM * 2);
  unsigned int*   pv  = (unsigned int*)alloc((size_t)DM * DM * 2);
  unsigned int*   po  = (unsigned int*)alloc((size_t)DM * DM * 2);
  unsigned int*   pfc = (unsigned int*)alloc((size_t)DM * FF * 2);
  unsigned int*   ppr = (unsigned int*)alloc((size_t)FF * DM * 2);

  const int ATTN_LDS = 16 * TT * 4 + 16 * TT * 2;   // 96 KB dynamic LDS
  hipFuncSetAttribute((const void*)attn_kernel,
                      hipFuncAttributeMaxDynamicSharedMemorySize, ATTN_LDS);

  auto gemm_blocks = [](int m, int n) { return ((m / 32) * (n / 64) + 7) / 8; };
  auto pack_blocks = [](int k, int n) { return ((k / 32) * (n / 16) + 7) / 8; };

  {
    int n = M * DM / 4;
    embed_kernel<<<(n + 255) / 256, 256, 0, stream>>>(idx, wte, wpe, xf);
  }

  for (int i = 0; i < NL; ++i) {
    const float* wq_i  = wq  + (size_t)i * DM * DM;
    const float* wk_i  = wk  + (size_t)i * DM * DM;
    const float* wv_i  = wv  + (size_t)i * DM * DM;
    const float* wo_i  = wo  + (size_t)i * DM * DM;
    const float* wfc_i = wfc + (size_t)i * DM * FF;
    const float* wpr_i = wpr + (size_t)i * FF * DM;

    ln_kernel<<<M, 256, 0, stream>>>(xf, ln1g + (size_t)i * DM, hb, DM);

    pack_kernel<<<pack_blocks(DM, DM), 256, 0, stream>>>(wq_i, pq, DM, DM);
    pack_kernel<<<pack_blocks(DM, DM), 256, 0, stream>>>(wk_i, pk, DM, DM);
    pack_kernel<<<pack_blocks(DM, DM), 256, 0, stream>>>(wv_i, pv, DM, DM);
    pack_kernel<<<pack_blocks(DM, DM), 256, 0, stream>>>(wo_i, po, DM, DM);

    gemm_wmma<1><<<gemm_blocks(M, DM), 256, 0, stream>>>(hb, pq, qb, nullptr, M, DM, DM);
    gemm_wmma<1><<<gemm_blocks(M, DM), 256, 0, stream>>>(hb, pk, kb, nullptr, M, DM, DM);
    gemm_wmma<1><<<gemm_blocks(M, DM), 256, 0, stream>>>(hb, pv, vb, nullptr, M, DM, DM);

    {
      int n = BSZ * NH * HDM * TT / 2;
      vtrans_kernel<<<(n + 255) / 256, 256, 0, stream>>>(vb, vt);
    }

    attn_kernel<<<BSZ * NH * (TT / 16), 32, ATTN_LDS, stream>>>(
        qb, kb, (const unsigned short*)vt, cxb);

    gemm_wmma<0><<<gemm_blocks(M, DM), 256, 0, stream>>>(cxb, po, xf, xf, M, DM, DM);

    ln_kernel<<<M, 256, 0, stream>>>(xf, ln2g + (size_t)i * DM, hb, DM);

    pack_kernel<<<pack_blocks(DM, FF), 256, 0, stream>>>(wfc_i, pfc, DM, FF);
    pack_kernel<<<pack_blocks(FF, DM), 256, 0, stream>>>(wpr_i, ppr, FF, DM);
    gemm_wmma<2><<<gemm_blocks(M, FF), 256, 0, stream>>>(hb, pfc, fcb, nullptr, M, FF, DM);
    gemm_wmma<0><<<gemm_blocks(M, DM), 256, 0, stream>>>(fcb, ppr, xf, xf, M, DM, FF);
  }

  ln_kernel<<<M, 256, 0, stream>>>(xf, lnfg, hb, DM);
  {
    int NT = (NV + 15) / 16;
    lmhead_kernel<<<(NT + 7) / 8, 256, 0, stream>>>(hb, wte, out);
  }
}
